// GASADecoder_88158498717790
// MI455X (gfx1250) — matrix operand order repaired and checked by reading, not compiled, not run on hardware
//
#include <hip/hip_runtime.h>
#include <hip/hip_bf16.h>

typedef __attribute__((ext_vector_type(16))) _Float16 v16h;
typedef __attribute__((ext_vector_type(8)))  _Float16 v8h;
typedef __attribute__((ext_vector_type(8)))  float    v8f;

#define D_       256
#define H_       8
#define HD_      32
#define L_       6
#define NQ_      100
#define QL_      101
#define DFF_     2048
#define B_       8
#define N_       4096

// GEMM macro-tile config
#define BM 64
#define BN 64
#define BK 32
#define APF    36   // floats per LDS row (A fp32: [BM][APF]) -- 144B, 16B-aligned, bank-skewed
#define BPITCH 40   // halves per LDS row (B transposed f16: [BN][BPITCH]) -- 80B

// ---------------------------------------------------------------------------
// WMMA GEMM: C[M,N] = act(A[M,K] @ W[K,N] + bias [+ resid])
// Block = 256 threads (8 waves) computes a 64x64 C macro-tile.
// Wave (wm 0..3, wn 0..1) owns a 16x32 strip: one A fragment, two B fragments,
// two V_WMMA_F32_16X16X32_F16 per K slab.
//
// A tile staging uses the CDNA5 async copy path: GLOBAL_LOAD_ASYNC_TO_LDS_B128
// (ASYNCcnt-tracked DMA, no VGPR round trip), fp32 image in LDS, converted to
// f16 at fragment-read time. B tile is loaded with coalesced float4 reads,
// converted, and stored transposed [n][k] in f16 so each fragment read is an
// aligned ds_load_b128. Fragment layouts used here:
//   A 16x32 f16 : lane L (m=L&15, half=L>>4): elem e -> k=half*16+e
//   B 32x16 f16 : lane L (n=L&15, half=L>>4): elem e -> k=half*16+e
//   C 16x16 f32 : lane L (n=L&15), VGPR r -> m=(L>>4)*8+r
// Rows m >= M are clamped (never zeroed): they only feed C rows that the
// epilogue never stores, so no divergent masking is needed anywhere.
// ---------------------------------------------------------------------------
__global__ __launch_bounds__(256)
void k_gemm_wmma(const float* __restrict__ A, const float* __restrict__ W,
                 const float* __restrict__ bias, const float* __restrict__ resid,
                 float* __restrict__ C, int M, int K, int N, int relu, int nblocks) {
  __shared__ __align__(16) float    As[BM * APF];
  __shared__ __align__(16) _Float16 Bs[BN * BPITCH];

  int bm = blockIdx.x / nblocks;
  int bn = blockIdx.x % nblocks;
  int mbase = bm * BM;
  int nbase = bn * BN;

  int t    = threadIdx.x;
  int lane = t & 31;
  int wave = t >> 5;        // 0..7
  int half = lane >> 4;
  int l15  = lane & 15;
  int wm   = wave >> 1;     // 0..3 : 16-row band
  int wn   = wave & 1;      // 0..1 : 32-col band

  // staging coordinates
  int a_row  = t >> 2;          // 0..63
  int a_col8 = (t & 3) * 8;     // 0,8,16,24 (floats; two 16B chunks)
  int b_krow = t >> 3;          // 0..31
  int b_col8 = (t & 7) * 8;     // 0..56

  int a_row_c = mbase + a_row;
  if (a_row_c > M - 1) a_row_c = M - 1;      // clamp; no zeroing needed
  unsigned a_lds = (unsigned)(size_t)(&As[a_row * APF + a_col8]);

  v8f acc0 = {};
  v8f acc1 = {};

  for (int kb = 0; kb < K; kb += BK) {
    // ---- stage A (BM x BK fp32) via async global->LDS DMA (2 x 16B/thread) ----
    {
      const float* gsrc = A + (size_t)a_row_c * K + kb + a_col8;
      asm volatile("global_load_async_to_lds_b128 %0, %1, off\n\t"
                   "global_load_async_to_lds_b128 %0, %1, off offset:16"
                   :
                   : "v"(a_lds), "v"(gsrc)
                   : "memory");
    }
    // ---- stage B (BK x BN fp32 -> transposed f16 LDS) : 2 x float4 per thread ----
    {
      const float4* src = (const float4*)(W + (size_t)(kb + b_krow) * N + nbase + b_col8);
      float4 x0 = src[0];
      float4 x1 = src[1];
      Bs[(b_col8 + 0) * BPITCH + b_krow] = (_Float16)x0.x;
      Bs[(b_col8 + 1) * BPITCH + b_krow] = (_Float16)x0.y;
      Bs[(b_col8 + 2) * BPITCH + b_krow] = (_Float16)x0.z;
      Bs[(b_col8 + 3) * BPITCH + b_krow] = (_Float16)x0.w;
      Bs[(b_col8 + 4) * BPITCH + b_krow] = (_Float16)x1.x;
      Bs[(b_col8 + 5) * BPITCH + b_krow] = (_Float16)x1.y;
      Bs[(b_col8 + 6) * BPITCH + b_krow] = (_Float16)x1.z;
      Bs[(b_col8 + 7) * BPITCH + b_krow] = (_Float16)x1.w;
    }
    if (kb + BK < K) {   // prefetch next W slab (global_prefetch_b8 hint)
      __builtin_prefetch(W + (size_t)(kb + BK + b_krow) * N + nbase + b_col8, 0, 1);
    }
    // wait for this wave's async DMA, then workgroup barrier
    asm volatile("s_wait_asynccnt 0x0" ::: "memory");
    __syncthreads();

    // ---- A fragment: fp32 from LDS (4 x ds_load_b128) + cvt to f16 ----
    const float* ap = &As[(wm * 16 + l15) * APF];
    float4 al0 = *(const float4*)(ap + half * 16);
    float4 al1 = *(const float4*)(ap + half * 16 + 4);
    float4 ah0 = *(const float4*)(ap + half * 16 + 8);
    float4 ah1 = *(const float4*)(ap + half * 16 + 12);
    v16h a;
    a[0]  = (_Float16)al0.x; a[1]  = (_Float16)al0.y;
    a[2]  = (_Float16)al0.z; a[3]  = (_Float16)al0.w;
    a[4]  = (_Float16)al1.x; a[5]  = (_Float16)al1.y;
    a[6]  = (_Float16)al1.z; a[7]  = (_Float16)al1.w;
    a[8]  = (_Float16)ah0.x; a[9]  = (_Float16)ah0.y;
    a[10] = (_Float16)ah0.z; a[11] = (_Float16)ah0.w;
    a[12] = (_Float16)ah1.x; a[13] = (_Float16)ah1.y;
    a[14] = (_Float16)ah1.z; a[15] = (_Float16)ah1.w;

    // ---- B fragments: f16 from transposed LDS (4 x ds_load_b128) ----
    const _Float16* bp0 = &Bs[(wn * 32 + l15) * BPITCH + half * 16];
    const _Float16* bp1 = &Bs[(wn * 32 + 16 + l15) * BPITCH + half * 16];
    v8h b0lo = *(const v8h*)(bp0);
    v8h b0hi = *(const v8h*)(bp0 + 8);
    v8h b1lo = *(const v8h*)(bp1);
    v8h b1hi = *(const v8h*)(bp1 + 8);
    v16h b0, b1;
#pragma unroll
    for (int e = 0; e < 8; ++e) {
      b0[e] = b0lo[e]; b0[8 + e] = b0hi[e];
      b1[e] = b1lo[e]; b1[8 + e] = b1hi[e];
    }
    acc0 = __builtin_amdgcn_wmma_f32_16x16x32_f16(false, a, false, b0,
                                                  (short)0, acc0, false, false);
    acc1 = __builtin_amdgcn_wmma_f32_16x16x32_f16(false, a, false, b1,
                                                  (short)0, acc1, false, false);
    __syncthreads();
  }

  // ---- epilogue ----
#pragma unroll
  for (int r = 0; r < 8; ++r) {
    int mm  = mbase + wm * 16 + half * 8 + r;
    int nn0 = nbase + wn * 32 + l15;
    int nn1 = nn0 + 16;
    if (mm < M) {
      float v0 = acc0[r];
      float v1 = acc1[r];
      if (bias)  { v0 += bias[nn0]; v1 += bias[nn1]; }
      if (resid) { v0 += resid[(size_t)mm * N + nn0]; v1 += resid[(size_t)mm * N + nn1]; }
      if (relu)  { v0 = v0 > 0.f ? v0 : 0.f; v1 = v1 > 0.f ? v1 : 0.f; }
      C[(size_t)mm * N + nn0] = v0;
      C[(size_t)mm * N + nn1] = v1;
    }
  }
}

// ---------------------------------------------------------------------------
// Flash-style attention: one wave per (b,h,q). lane = head channel (HD==32).
// Online softmax over KL keys; optional distance bias for cross-attention.
// ---------------------------------------------------------------------------
__global__ void k_attn(const float* __restrict__ Q, const float* __restrict__ Kt,
                       const float* __restrict__ V, float* __restrict__ O,
                       int B, int QL, int KL, float scale,
                       const float* __restrict__ posq,   // B*QL*3 or null
                       const float* __restrict__ pm,     // B*KL*3 or null
                       const float* __restrict__ beta_l) // H floats or null
{
  int wave = blockIdx.x * (blockDim.x >> 5) + (threadIdx.x >> 5);
  int total = B * H_ * QL;
  if (wave >= total) return;
  int q = wave % QL;
  int t = wave / QL;
  int h = t % H_;
  int b = t / H_;
  int lane = threadIdx.x & 31;

  float ql = Q[((size_t)(b * QL + q)) * D_ + h * HD_ + lane];
  bool use_bias = (posq != nullptr);
  float nb = 0.f, p0 = 0.f, p1 = 0.f, p2 = 0.f;
  if (use_bias) {
    float be = beta_l[h];
    nb = log1pf(__expf(be));            // softplus
    p0 = posq[(size_t)(b * QL + q) * 3 + 0];
    p1 = posq[(size_t)(b * QL + q) * 3 + 1];
    p2 = posq[(size_t)(b * QL + q) * 3 + 2];
  }
  float m_run = -3.0e38f, l_run = 0.f, acc = 0.f;
  for (int n = 0; n < KL; ++n) {
    float kv = Kt[((size_t)(b * KL + n)) * D_ + h * HD_ + lane];
    float s = ql * kv;
#pragma unroll
    for (int off = 16; off; off >>= 1) s += __shfl_xor(s, off, 32);
    s *= scale;
    if (use_bias) {
      float d0 = p0 - pm[(size_t)(b * KL + n) * 3 + 0];
      float d1 = p1 - pm[(size_t)(b * KL + n) * 3 + 1];
      float d2 = p2 - pm[(size_t)(b * KL + n) * 3 + 2];
      s -= nb * (d0 * d0 + d1 * d1 + d2 * d2);
    }
    float mn   = fmaxf(m_run, s);
    float corr = __expf(m_run - mn);
    float ex   = __expf(s - mn);
    float vv   = V[((size_t)(b * KL + n)) * D_ + h * HD_ + lane];
    l_run = l_run * corr + ex;
    acc   = acc  * corr + ex * vv;
    m_run = mn;
  }
  O[((size_t)(b * QL + q)) * D_ + h * HD_ + lane] = acc / l_run;
}

// ---------------------------------------------------------------------------
// LayerNorm over D=256, one wave per row, 8 elements per lane.
// ---------------------------------------------------------------------------
__global__ void k_ln(const float* __restrict__ x, const float* __restrict__ g,
                     const float* __restrict__ b, float* __restrict__ y, int rows) {
  int row = blockIdx.x * (blockDim.x >> 5) + (threadIdx.x >> 5);
  if (row >= rows) return;
  int lane = threadIdx.x & 31;
  const float* xr = x + (size_t)row * D_;
  float v[8];
  float s = 0.f;
#pragma unroll
  for (int i = 0; i < 8; ++i) { v[i] = xr[lane + 32 * i]; s += v[i]; }
#pragma unroll
  for (int off = 16; off; off >>= 1) s += __shfl_xor(s, off, 32);
  float mean = s * (1.0f / D_);
  float var = 0.f;
#pragma unroll
  for (int i = 0; i < 8; ++i) { float d = v[i] - mean; var += d * d; }
#pragma unroll
  for (int off = 16; off; off >>= 1) var += __shfl_xor(var, off, 32);
  var *= (1.0f / D_);
  float rs = rsqrtf(var + 1e-5f);
  float* yr = y + (size_t)row * D_;
#pragma unroll
  for (int i = 0; i < 8; ++i) {
    int j = lane + 32 * i;
    yr[j] = (v[i] - mean) * rs * g[j] + b[j];
  }
}

// ---------------------------------------------------------------------------
// Positional encoding: mem = image + enc @ wpe_W + wpe_b.
// FREQS = geomspace(1,512,10) == exact powers of two (step 2x).
// One block (256 threads) per (b,n) row; enc[60] built in LDS.
// ---------------------------------------------------------------------------
__global__ void k_posenc(const float* __restrict__ img, const float* __restrict__ pm,
                         const float* __restrict__ wpeW, const float* __restrict__ wpeb,
                         float* __restrict__ mem) {
  __shared__ float enc[60];
  size_t row = blockIdx.x;
  int j = threadIdx.x;
  if (j < 60) {
    int c = j / 20;
    int f = j % 20;
    float freq = (float)(1 << (f % 10));
    float ang = pm[row * 3 + c] * freq;
    enc[j] = (f < 10) ? sinf(ang) : cosf(ang);
  }
  __syncthreads();
  float s = wpeb[j];
#pragma unroll 4
  for (int t = 0; t < 60; ++t) s += enc[t] * wpeW[t * D_ + j];
  mem[row * D_ + j] = img[row * D_ + j] + s;
}

// pos_q = x @ pos_W (256x3) + pos_b, naive: one thread per output scalar.
__global__ void k_posq(const float* __restrict__ x, const float* __restrict__ W,
                       const float* __restrict__ b3, float* __restrict__ pq, int rows) {
  int idx = blockIdx.x * blockDim.x + threadIdx.x;
  if (idx >= rows * 3) return;
  int row = idx / 3, c = idx % 3;
  float s = b3[c];
  for (int k = 0; k < D_; ++k) s += x[(size_t)row * D_ + k] * W[k * 3 + c];
  pq[idx] = s;
}

__global__ void k_init_x(const float* __restrict__ qe, const float* __restrict__ pt,
                         float* __restrict__ x) {
  int idx = blockIdx.x * blockDim.x + threadIdx.x;
  if (idx >= B_ * QL_ * D_) return;
  int j = idx % D_;
  int q = (idx / D_) % QL_;
  x[idx] = (q < NQ_) ? qe[q * D_ + j] : pt[j];
}

__global__ void k_copy_hs(const float* __restrict__ x, float* __restrict__ out) {
  int idx = blockIdx.x * blockDim.x + threadIdx.x;
  if (idx >= B_ * NQ_ * D_) return;
  int b = idx / (NQ_ * D_);
  int rem = idx % (NQ_ * D_);
  out[idx] = x[(size_t)b * QL_ * D_ + rem];
}

__global__ void k_gather_pres(const float* __restrict__ x, float* __restrict__ pr) {
  int idx = blockIdx.x * blockDim.x + threadIdx.x;
  if (idx >= B_ * D_) return;
  int b = idx / D_, j = idx % D_;
  pr[idx] = x[((size_t)b * QL_ + NQ_) * D_ + j];
}

// logit = h(8x256) @ pW3(256x1) + pb3 : one wave per batch row.
__global__ void k_logit(const float* __restrict__ h, const float* __restrict__ w,
                        const float* __restrict__ b, float* __restrict__ out) {
  int bi = blockIdx.x;
  int lane = threadIdx.x;
  float s = 0.f;
  for (int i = lane; i < D_; i += 32) s += h[bi * D_ + i] * w[i];
#pragma unroll
  for (int off = 16; off; off >>= 1) s += __shfl_xor(s, off, 32);
  if (lane == 0) out[bi] = b[0] + s;
}

// ---------------------------------------------------------------------------
// host-side launchers
// ---------------------------------------------------------------------------
static inline void gemm(const float* A, const float* W, const float* bias,
                        const float* resid, float* C, int M, int K, int N,
                        int relu, hipStream_t s) {
  int mb = (M + BM - 1) / BM;
  int nb = N / BN;
  k_gemm_wmma<<<mb * nb, 256, 0, s>>>(A, W, bias, resid, C, M, K, N, relu, nb);
}

static inline void attn(const float* Q, const float* K, const float* V, float* O,
                        int B, int QL, int KL, const float* posq, const float* pm,
                        const float* beta_l, hipStream_t s) {
  float scale = 0.17677669529663687f; // 1/sqrt(32)
  int waves = B * H_ * QL;
  int blocks = (waves + 7) / 8;
  k_attn<<<blocks, 256, 0, s>>>(Q, K, V, O, B, QL, KL, scale, posq, pm, beta_l);
}

static inline void lnorm(const float* x, const float* g, const float* b, float* y,
                         int rows, hipStream_t s) {
  int blocks = (rows + 7) / 8;
  k_ln<<<blocks, 256, 0, s>>>(x, g, b, y, rows);
}

extern "C" void kernel_launch(void* const* d_in, const int* in_sizes, int n_in,
                              void* d_out, int out_size, void* d_ws, size_t ws_size,
                              hipStream_t stream) {
  (void)in_sizes; (void)n_in; (void)out_size; (void)ws_size;
  const float* image = (const float*)d_in[0];
  const float* text  = (const float*)d_in[1];
  const float* pm    = (const float*)d_in[2];
  const float* qe    = (const float*)d_in[3];
  const float* ptok  = (const float*)d_in[4];
  const float* tpW   = (const float*)d_in[5];
  const float* tpb   = (const float*)d_in[6];
  const float* tWq = (const float*)d_in[7],  *tbq = (const float*)d_in[8];
  const float* tWk = (const float*)d_in[9],  *tbk = (const float*)d_in[10];
  const float* tWv = (const float*)d_in[11], *tbv = (const float*)d_in[12];
  const float* tWo = (const float*)d_in[13], *tbo = (const float*)d_in[14];
  const float* tlng = (const float*)d_in[15], *tlnb = (const float*)d_in[16];
  const float* wpeW = (const float*)d_in[17], *wpeb = (const float*)d_in[18];
  const float* saWq = (const float*)d_in[19], *sabq = (const float*)d_in[20];
  const float* saWk = (const float*)d_in[21], *sabk = (const float*)d_in[22];
  const float* saWv = (const float*)d_in[23], *sabv = (const float*)d_in[24];
  const float* saWo = (const float*)d_in[25], *sabo = (const float*)d_in[26];
  const float* caWq = (const float*)d_in[27], *cabq = (const float*)d_in[28];
  const float* caWk = (const float*)d_in[29], *cabk = (const float*)d_in[30];
  const float* caWv = (const float*)d_in[31], *cabv = (const float*)d_in[32];
  const float* caWo = (const float*)d_in[33], *cabo = (const float*)d_in[34];
  const float* fW1 = (const float*)d_in[35], *fb1 = (const float*)d_in[36];
  const float* fW2 = (const float*)d_in[37], *fb2 = (const float*)d_in[38];
  const float* ln1g = (const float*)d_in[39], *ln1b = (const float*)d_in[40];
  const float* ln2g = (const float*)d_in[41], *ln2b = (const float*)d_in[42];
  const float* ln3g = (const float*)d_in[43], *ln3b = (const float*)d_in[44];
  const float* posW = (const float*)d_in[45], *posb = (const float*)d_in[46];
  const float* beta = (const float*)d_in[47];
  const float* fing = (const float*)d_in[48], *finb = (const float*)d_in[49];
  const float* pW1 = (const float*)d_in[50], *pb1 = (const float*)d_in[51];
  const float* pW2 = (const float*)d_in[52], *pb2 = (const float*)d_in[53];
  const float* pW3 = (const float*)d_in[54], *pb3 = (const float*)d_in[55];
  float* out = (float*)d_out;

  // ---- workspace carve (floats) ----
  float* ws = (float*)d_ws;
  float* mem = ws;                 ws += (size_t)B_ * N_ * D_;        // 8.39M
  float* Kb  = ws;                 ws += (size_t)B_ * N_ * D_;        // 8.39M
  float* Vb  = ws;                 ws += (size_t)B_ * N_ * D_;        // 8.39M
  float* Qb  = ws;                 ws += (size_t)B_ * QL_ * D_;
  float* Oh  = ws;                 ws += (size_t)B_ * QL_ * D_;
  float* xa  = ws;                 ws += (size_t)B_ * QL_ * D_;
  float* xb  = ws;                 ws += (size_t)B_ * QL_ * D_;
  float* txt = ws;                 ws += (size_t)B_ * 32 * D_;
  float* pq  = ws;                 ws += (size_t)B_ * QL_ * 4;
  float* hf  = ws;                 ws += (size_t)B_ * QL_ * DFF_;     // 1.65M
  float* pr1 = ws;                 ws += (size_t)B_ * D_;
  float* pr2 = ws;                 ws += (size_t)B_ * D_;

  const int Mx = B_ * QL_;   // 808 query-token rows
  const int Mm = B_ * N_;    // 32768 memory rows
  const int Mt = B_ * 32;    // 256 text rows

  // 1) positional encoding -> mem
  k_posenc<<<Mm, 256, 0, stream>>>(image, pm, wpeW, wpeb, mem);

  // 2) text projection
  gemm(text, tpW, tpb, nullptr, txt, Mt, D_, D_, 0, stream);

  // 3) init x (queries + presence token)
  {
    int n = Mx * D_;
    k_init_x<<<(n + 255) / 256, 256, 0, stream>>>(qe, ptok, xa);
  }

  // 4) text cross-attention block
  gemm(xa,  tWq, tbq, nullptr, Qb, Mx, D_, D_, 0, stream);
  gemm(txt, tWk, tbk, nullptr, Kb, Mt, D_, D_, 0, stream);
  gemm(txt, tWv, tbv, nullptr, Vb, Mt, D_, D_, 0, stream);
  attn(Qb, Kb, Vb, Oh, B_, QL_, 32, nullptr, nullptr, nullptr, stream);
  gemm(Oh, tWo, tbo, xa, xb, Mx, D_, D_, 0, stream);
  lnorm(xb, tlng, tlnb, xa, Mx, stream);

  // 5) decoder layers
  for (int i = 0; i < L_; ++i) {
    const size_t wofs = (size_t)i * D_ * D_;
    const size_t bofs = (size_t)i * D_;
    // --- self-attention ---
    gemm(xa, saWq + wofs, sabq + bofs, nullptr, Qb, Mx, D_, D_, 0, stream);
    gemm(xa, saWk + wofs, sabk + bofs, nullptr, Kb, Mx, D_, D_, 0, stream);
    gemm(xa, saWv + wofs, sabv + bofs, nullptr, Vb, Mx, D_, D_, 0, stream);
    attn(Qb, Kb, Vb, Oh, B_, QL_, QL_, nullptr, nullptr, nullptr, stream);
    gemm(Oh, saWo + wofs, sabo + bofs, xa, xb, Mx, D_, D_, 0, stream);
    lnorm(xb, ln1g + bofs, ln1b + bofs, xa, Mx, stream);
    // --- geometric bias query positions ---
    {
      int n = Mx * 3;
      k_posq<<<(n + 255) / 256, 256, 0, stream>>>(xa, posW + (size_t)i * D_ * 3,
                                                  posb + (size_t)i * 3, pq, Mx);
    }
    // --- cross-attention over memory (biased) ---
    gemm(xa,  caWq + wofs, cabq + bofs, nullptr, Qb, Mx, D_, D_, 0, stream);
    gemm(mem, caWk + wofs, cabk + bofs, nullptr, Kb, Mm, D_, D_, 0, stream);
    gemm(mem, caWv + wofs, cabv + bofs, nullptr, Vb, Mm, D_, D_, 0, stream);
    attn(Qb, Kb, Vb, Oh, B_, QL_, N_, pq, pm, beta + (size_t)i * H_, stream);
    gemm(Oh, caWo + wofs, cabo + bofs, xa, xb, Mx, D_, D_, 0, stream);
    lnorm(xb, ln2g + bofs, ln2b + bofs, xa, Mx, stream);
    // --- FFN ---
    gemm(xa, fW1 + (size_t)i * D_ * DFF_, fb1 + (size_t)i * DFF_, nullptr, hf,
         Mx, D_, DFF_, 1, stream);
    gemm(hf, fW2 + (size_t)i * DFF_ * D_, fb2 + bofs, xa, xb, Mx, DFF_, D_, 0, stream);
    lnorm(xb, ln3g + bofs, ln3b + bofs, xa, Mx, stream);
  }

  // 6) final LN + outputs
  lnorm(xa, fing, finb, xb, Mx, stream);
  {
    int n = B_ * NQ_ * D_;
    k_copy_hs<<<(n + 255) / 256, 256, 0, stream>>>(xb, out);
  }
  {
    int n = B_ * D_;
    k_gather_pres<<<(n + 255) / 256, 256, 0, stream>>>(xb, pr1);
  }
  gemm(pr1, pW1, pb1, nullptr, pr2, B_, D_, D_, 1, stream);
  gemm(pr2, pW2, pb2, nullptr, pr1, B_, D_, D_, 1, stream);
  k_logit<<<B_, 32, 0, stream>>>(pr1, pW3, pb3, out + (size_t)B_ * NQ_ * D_);
}
